// CVectorQuantiser_88811333747151
// MI455X (gfx1250) — matrix operand order repaired and checked by reading, not compiled, not run on hardware
//
#include <hip/hip_runtime.h>
#include <hip/hip_bf16.h>
#include <stdint.h>

// ---------------------------------------------------------------------------
// VQ codebook lookup on MI455X (gfx1250, wave32, WMMA + async LDS staging).
// z: [16,256,32,32] f32, weight: [8192,256] f32 (rows l2-normalized)
// out: z_q [16,256,32,32] f32  |  loss scalar f32  |  idx [16384] i32
// ---------------------------------------------------------------------------

typedef __attribute__((ext_vector_type(16))) __bf16 v16bf;
typedef __attribute__((ext_vector_type(8)))  float  v8f;

#define N_TOK   16384
#define DIM     256
#define K_CODES 8192
#define BETA    0.25f

union BFrag { v16bf v; uint4 u[2]; };

__device__ __forceinline__ uint16_t f32_to_bf16(float f) {
  uint32_t x = __float_as_uint(f);
  uint32_t r = x + 0x7FFFu + ((x >> 16) & 1u);   // round-to-nearest-even
  return (uint16_t)(r >> 16);
}

// ---------------------------------------------------------------------------
__global__ void vq_init(float* loss) { *loss = 0.0f; }

// weight f32 [8192,256] -> bf16, 4 elems/thread
__global__ void vq_convert_w(const float* __restrict__ w, uint16_t* __restrict__ wbf) {
  int i = (blockIdx.x * blockDim.x + threadIdx.x) * 4;
  #pragma unroll
  for (int j = 0; j < 4; ++j) wbf[i + j] = f32_to_bf16(w[i + j]);
}

// z [16,256,32,32] -> l2-normalized bf16 tokens [16384,256] + inv_norm [16384]
__global__ void vq_normalize(const float* __restrict__ z,
                             uint16_t* __restrict__ znorm,
                             float* __restrict__ inv_norm) {
  __shared__ float part[4][64];
  int tid = threadIdx.x;
  int hw_off = tid & 63, dg = tid >> 6;
  int token = blockIdx.x * 64 + hw_off;
  int b = token >> 10, hw = token & 1023;
  const float* zp = z + (size_t)b * (DIM * 1024) + hw;

  float ss = 0.0f;
  #pragma unroll 4
  for (int i = 0; i < 64; ++i) {
    int d = dg * 64 + i;
    float v = zp[(size_t)d * 1024];
    ss += v * v;
  }
  part[dg][hw_off] = ss;
  __syncthreads();
  float tot = part[0][hw_off] + part[1][hw_off] + part[2][hw_off] + part[3][hw_off];
  float inv = 1.0f / fmaxf(sqrtf(tot), 1e-12f);
  if (dg == 0) inv_norm[token] = inv;

  uint16_t* op = znorm + (size_t)token * DIM + dg * 64;
  #pragma unroll 4
  for (int i = 0; i < 64; ++i) {
    int d = dg * 64 + i;
    op[i] = f32_to_bf16(zp[(size_t)d * 1024] * inv);
  }
}

// ---------------------------------------------------------------------------
// Argmax GEMM: block = 256 thr (8 waves), 128 tokens/block, full codebook loop.
// Each wave: 16 tokens; A tile 16x256 bf16 in 8 v16bf register fragments.
// 32-code B tile (16 KB) double-buffered in LDS, staged asynchronously with
// global_load_async_to_lds_b128 (ASYNCcnt); one barrier per k-step.
// Per k-step: 2 x (16 ds_load_b128 -> sched_barrier -> 8 back-to-back WMMAs).
// ---------------------------------------------------------------------------
__global__ void __launch_bounds__(256, 1)
vq_argmax(const uint16_t* __restrict__ znorm,
          const uint16_t* __restrict__ wbf,
          int* __restrict__ idx_out) {
  __shared__ uint16_t Bt[2][32 * DIM];           // 2 x 16 KB double buffer

  int tid  = threadIdx.x;
  int wave = tid >> 5;
  int lane = tid & 31;
  int half = lane >> 4;                          // 0: K 0..7/16..23  1: K 8..15/24..31
  int n    = lane & 15;                          // N column / A row
  int tbase = blockIdx.x * 128 + wave * 16;

  // --- load A fragments (ISA 16-bit A 16x32 layout) ---
  BFrag A[8];
  const uint16_t* arow = znorm + (size_t)(tbase + n) * DIM;
  #pragma unroll
  for (int db = 0; db < 8; ++db) {
    int base = db * 32 + half * 8;
    A[db].u[0] = *(const uint4*)(arow + base);       // K local base..base+7
    A[db].u[1] = *(const uint4*)(arow + base + 16);  // K local base+16..+23
  }

  float best[8];
  int   bidx[8];
  #pragma unroll
  for (int j = 0; j < 8; ++j) { best[j] = -3.0e38f; bidx[j] = 0; }

  // staging coords: 256 threads x 64 B = one 16 KB tile (32 codes x 256 dims)
  int srow = tid >> 3;                 // code row 0..31
  int scol = (tid & 7) * 32;           // 32 bf16 (64 B) per thread
  const uint16_t* wsrc = wbf + (size_t)srow * DIM + scol;
  uint32_t lds0 = (uint32_t)(uintptr_t)&Bt[0][srow * DIM + scol];
  uint32_t lds1 = (uint32_t)(uintptr_t)&Bt[1][srow * DIM + scol];

  // prologue: async-stage tile 0 into buffer 0
  {
    uint64_t g = (uint64_t)(uintptr_t)wsrc;
    asm volatile("global_load_async_to_lds_b128 %0, %1, off"
                 :: "v"(lds0), "v"(g) : "memory");
    asm volatile("global_load_async_to_lds_b128 %0, %1, off offset:16"
                 :: "v"(lds0), "v"(g) : "memory");
    asm volatile("global_load_async_to_lds_b128 %0, %1, off offset:32"
                 :: "v"(lds0), "v"(g) : "memory");
    asm volatile("global_load_async_to_lds_b128 %0, %1, off offset:48"
                 :: "v"(lds0), "v"(g) : "memory");
    asm volatile("s_wait_asynccnt 0x0" ::: "memory");
  }
  __syncthreads();

  for (int k0 = 0; k0 < K_CODES; k0 += 32) {
    int cur = (k0 >> 5) & 1;

    // async-stage next 16 KB tile into the other buffer while we compute
    if (k0 + 32 < K_CODES) {
      uint64_t g = (uint64_t)(uintptr_t)(wsrc + (size_t)(k0 + 32) * DIM);
      uint32_t dst = cur ? lds0 : lds1;
      asm volatile("global_load_async_to_lds_b128 %0, %1, off"
                   :: "v"(dst), "v"(g) : "memory");
      asm volatile("global_load_async_to_lds_b128 %0, %1, off offset:16"
                   :: "v"(dst), "v"(g) : "memory");
      asm volatile("global_load_async_to_lds_b128 %0, %1, off offset:32"
                   :: "v"(dst), "v"(g) : "memory");
      asm volatile("global_load_async_to_lds_b128 %0, %1, off offset:48"
                   :: "v"(dst), "v"(g) : "memory");
    }

    // two 16-code sub-tiles: 16 ds loads -> fence -> 8 back-to-back WMMAs
    #pragma unroll
    for (int g = 0; g < 2; ++g) {
      const uint16_t* bbase = &Bt[cur][(g * 16 + n) * DIM + half * 8];
      BFrag B[8];
      #pragma unroll
      for (int db = 0; db < 8; ++db) {
        B[db].u[0] = *(const uint4*)(bbase + db * 32);
        B[db].u[1] = *(const uint4*)(bbase + db * 32 + 16);
      }
      __builtin_amdgcn_sched_barrier(0);   // keep all 8 frags live: 1 wait, 8 WMMAs
      v8f acc = {};
      #pragma unroll
      for (int db = 0; db < 8; ++db)
        acc = __builtin_amdgcn_wmma_f32_16x16x32_bf16(
            false, A[db].v, false, B[db].v, (short)0, acc, false, false);
      __builtin_amdgcn_sched_barrier(0);

      int code = k0 + g * 16 + n;        // this lane's N column
      #pragma unroll
      for (int j = 0; j < 8; ++j)        // VGPR j: M = j + 8*half
        if (acc[j] > best[j]) { best[j] = acc[j]; bidx[j] = code; }
    }

    asm volatile("s_wait_asynccnt 0x0" ::: "memory");
    __syncthreads();                     // next tile staged; old buffer reusable
  }

  // cross-lane argmax over the 16 N-lanes (within each half), min-index ties
  #pragma unroll
  for (int j = 0; j < 8; ++j) {
    float v = best[j];
    int   id = bidx[j];
    #pragma unroll
    for (int off = 8; off > 0; off >>= 1) {
      float ov = __shfl_xor(v, off, 16);
      int   oi = __shfl_xor(id, off, 16);
      if (ov > v || (ov == v && oi < id)) { v = ov; id = oi; }
    }
    if (n == 0) idx_out[tbase + j + 8 * half] = id;
  }
}

// ---------------------------------------------------------------------------
// Gather z_q = weight[idx] back to [B,C,H,W]; exact fp32 commitment loss.
// ---------------------------------------------------------------------------
__global__ void vq_gather(const float* __restrict__ z,
                          const float* __restrict__ weight,
                          const float* __restrict__ inv_norm,
                          const int* __restrict__ idx,
                          float* __restrict__ zq,
                          float* __restrict__ loss) {
  __shared__ float red[256];
  int tid = threadIdx.x;
  int hw_off = tid & 63, dg = tid >> 6;
  int token = blockIdx.x * 64 + hw_off;
  int b = token >> 10, hw = token & 1023;

  int id = idx[token];
  float inv = inv_norm[token];
  const float* zp   = z  + (size_t)b * (DIM * 1024) + hw;
  float*       op   = zq + (size_t)b * (DIM * 1024) + hw;
  const float* wrow = weight + (size_t)id * DIM;

  float s = 0.0f;
  #pragma unroll 4
  for (int i = 0; i < 64; ++i) {
    int d = dg * 64 + i;
    float wq = wrow[d];
    float zc = zp[(size_t)d * 1024] * inv;
    float df = wq - zc;
    s += df * df;
    op[(size_t)d * 1024] = wq;                    // STE forward value = codebook row
  }
  red[tid] = s;
  __syncthreads();
  for (int st = 128; st > 0; st >>= 1) {
    if (tid < st) red[tid] += red[tid + st];
    __syncthreads();
  }
  if (tid == 0)
    atomicAdd(loss, red[0] * (BETA / (float)(N_TOK * DIM)));
}

// ---------------------------------------------------------------------------
extern "C" void kernel_launch(void* const* d_in, const int* in_sizes, int n_in,
                              void* d_out, int out_size, void* d_ws, size_t ws_size,
                              hipStream_t stream) {
  const float* z      = (const float*)d_in[0];   // 16*256*32*32
  const float* weight = (const float*)d_in[1];   // 8192*256

  float* out   = (float*)d_out;
  float* zq    = out;                                   // 4,194,304 floats
  float* loss  = out + (size_t)N_TOK * DIM;             // 1 float
  int*   idxo  = (int*)(out + (size_t)N_TOK * DIM + 1); // 16384 ints

  uint16_t* wbf   = (uint16_t*)d_ws;                       // 4 MB
  uint16_t* znorm = wbf + (size_t)K_CODES * DIM;           // 8 MB
  float*    invn  = (float*)(znorm + (size_t)N_TOK * DIM); // 64 KB

  vq_init<<<1, 1, 0, stream>>>(loss);
  vq_convert_w<<<(K_CODES * DIM) / (256 * 4), 256, 0, stream>>>(weight, wbf);
  vq_normalize<<<N_TOK / 64, 256, 0, stream>>>(z, znorm, invn);
  vq_argmax<<<N_TOK / 128, 256, 0, stream>>>(znorm, wbf, idxo);
  vq_gather<<<N_TOK / 64, 256, 0, stream>>>(z, weight, invn, idxo, zq, loss);
}
